// Directionalmamba_7086696038787
// MI455X (gfx1250) — matrix operand order repaired and checked by reading, not compiled, not run on hardware
//
#include <hip/hip_runtime.h>
#include <stdint.h>

// ---------------- problem constants ----------------
#define BQ   8        // batch
#define CHN  256      // channels of xc
#define KCAT 512      // concat input channels
#define HH   64
#define WWD  64
#define HW   4096     // H*W
#define DD   64       // D = CH/4
#define EE   128      // E = 2D
#define NN   16       // state dim
#define LL   4096     // sequence length
#define SS   32       // B*4 sequences
#define JZ   256      // 2E
#define JX   48       // padded DTR+2N (36 -> 48)
#define EPS  1e-5f

typedef __bf16 bf16;
typedef __attribute__((ext_vector_type(16))) __bf16 v16bf;
typedef __attribute__((ext_vector_type(8)))  __bf16 v8bf;
typedef __attribute__((ext_vector_type(8)))  float  v8f;

// ---------------- WMMA fragment helpers (CDNA5 wave32 layouts) ----------------
__device__ __forceinline__ v16bf frag_cat(v8bf lo, v8bf hi) {
  return __builtin_shufflevector(lo, hi, 0,1,2,3,4,5,6,7,8,9,10,11,12,13,14,15);
}

// A: 16x32 bf16 tile from row-major MxK (K contiguous). lane m=l&15, hs=l>>4.
// VGPR0..3 hold K = hs*8 .. hs*8+7 ; VGPR4..7 hold K = 16+hs*8 .. 16+hs*8+7
__device__ __forceinline__ v16bf load_fragA(const bf16* __restrict__ A, int lda,
                                            int row0, int k0, int lane) {
  int m  = lane & 15;
  int hs = (lane >> 4) & 1;
  const bf16* p = A + (size_t)(row0 + m) * lda + k0 + hs * 8;
  v8bf lo = *(const v8bf*)(p);
  v8bf hi = *(const v8bf*)(p + 16);
  return frag_cat(lo, hi);
}

// B: 32x16 bf16 tile, stored as B^T row-major (N rows x K cols, K contiguous).
// lane n=l&15, hs=l>>4; VGPR0..3 hold K = hs*8 .. hs*8+7 ; VGPR4..7 hold K = 16+hs*8 .. 16+hs*8+7
__device__ __forceinline__ v16bf load_fragB(const bf16* __restrict__ Bt, int ldb,
                                            int n0, int k0, int lane) {
  int n  = lane & 15;
  int hs = (lane >> 4) & 1;
  const bf16* p = Bt + (size_t)(n0 + n) * ldb + k0 + hs * 8;
  v8bf lo = *(const v8bf*)(p);
  v8bf hi = *(const v8bf*)(p + 16);
  return frag_cat(lo, hi);
}

__device__ __forceinline__ v8f wmma_bf16(v16bf a, v16bf b, v8f c) {
  // D = A x B + C, f32 accumulate; emits v_wmma_f32_16x16x32_bf16
  return __builtin_amdgcn_wmma_f32_16x16x32_bf16(false, a, false, b, (short)0, c,
                                                 false, false);
}

__device__ __forceinline__ float sigmoidf_(float x) { return 1.f / (1.f + __expf(-x)); }
__device__ __forceinline__ float siluf_(float x)    { return x * sigmoidf_(x); }
__device__ __forceinline__ float softplusf_(float x) {
  return (x > 20.f) ? x : log1pf(__expf(x));
}

// ---------------- K0: convert / pad weights to bf16 ----------------
__global__ void k_prep_weights(const float* __restrict__ w1, const float* __restrict__ Win,
                               const float* __restrict__ Wx, const float* __restrict__ Wout,
                               bf16* __restrict__ w1b, bf16* __restrict__ winb,
                               bf16* __restrict__ wxpb, bf16* __restrict__ woutb) {
  int i = blockIdx.x * blockDim.x + threadIdx.x;
  if (i < CHN * KCAT) w1b[i] = (bf16)w1[i];                       // 256x512
  if (i < JZ * DD)    winb[i] = (bf16)Win[i];                     // 256x64
  if (i < JX * EE) {                                              // 48x128 (zero-pad rows 36..47)
    int j = i >> 7, e = i & 127;
    wxpb[i] = (j < 36) ? (bf16)Wx[j * EE + e] : (bf16)0.f;
  }
  if (i < DD * EE)    woutb[i] = (bf16)Wout[i];                   // 64x128
}

// ---------------- K1: concat + transpose to (b, p, ci) bf16 ----------------
__global__ void k_cat_transpose(const float* __restrict__ x, const float* __restrict__ x2,
                                bf16* __restrict__ cat) {
  int idx = blockIdx.x * blockDim.x + threadIdx.x;   // (b, c, p), p fastest -> coalesced reads
  int p = idx & (HW - 1);
  int c = (idx >> 12) & (KCAT - 1);
  int b = idx >> 21;
  float v = (c < CHN) ? x[(((size_t)b * CHN + c) << 12) + p]
                      : x2[(((size_t)b * CHN + (c - CHN)) << 12) + p];
  cat[(((size_t)b << 12) + p) * KCAT + c] = (bf16)v;
}

// ---------------- K2: GEMM1  xc = relu(bn(W1 x cat)) ----------------
// per-wave 64(co) x 16(p) tile, K=512: 4 WMMAs share one activation fragment
// (4x reduction in traffic on the large operand).  A = w1b (256x512),
// B^T = cat rows (p x 512).
__global__ void k_gemm1(const bf16* __restrict__ W, const bf16* __restrict__ Xp,
                        const float* __restrict__ b1, const float* __restrict__ bng,
                        const float* __restrict__ bnb, const float* __restrict__ bnm,
                        const float* __restrict__ bnv, float* __restrict__ xc) {
  int wave = (blockIdx.x * blockDim.x + threadIdx.x) >> 5;
  int lane = threadIdx.x & 31;
  int pt  = wave & 255;
  int cot = (wave >> 8) & 3;
  int b   = wave >> 10;
  int co0 = cot * 64, p0 = pt * 16;
  const bf16* Brow = Xp + ((size_t)b << 12) * KCAT;
  v8f acc0 = {}, acc1 = {}, acc2 = {}, acc3 = {};
  #pragma unroll 4
  for (int k = 0; k < KCAT; k += 32) {
    __builtin_prefetch(Brow + (size_t)p0 * KCAT + k + 128, 0, 0);  // global_prefetch_b8
    v16bf bb = load_fragB(Brow, KCAT, p0, k, lane);
    v16bf a0 = load_fragA(W, KCAT, co0 +  0, k, lane);
    v16bf a1 = load_fragA(W, KCAT, co0 + 16, k, lane);
    v16bf a2 = load_fragA(W, KCAT, co0 + 32, k, lane);
    v16bf a3 = load_fragA(W, KCAT, co0 + 48, k, lane);
    acc0 = wmma_bf16(a0, bb, acc0);
    acc1 = wmma_bf16(a1, bb, acc1);
    acc2 = wmma_bf16(a2, bb, acc2);
    acc3 = wmma_bf16(a3, bb, acc3);
  }
  int n = lane & 15, hs = lane >> 4;
  int pcol = p0 + n;
  v8f accs[4] = {acc0, acc1, acc2, acc3};
  #pragma unroll
  for (int t = 0; t < 4; t++) {
    #pragma unroll
    for (int v = 0; v < 8; v++) {
      int co = co0 + t * 16 + v + 8 * hs;
      float alpha = bng[co] * rsqrtf(bnv[co] + EPS);
      float beta  = (b1[co] - bnm[co]) * alpha + bnb[co];
      xc[(((size_t)b * CHN + co) << 12) + pcol] = fmaxf(accs[t][v] * alpha + beta, 0.f);
    }
  }
}

// ---------------- K3: directional rearrange -> xs (s, l, d) bf16 ----------------
__global__ void k_rearrange(const float* __restrict__ xc, bf16* __restrict__ xs) {
  int idx = blockIdx.x * blockDim.x + threadIdx.x;   // (s, l, d), d fastest
  int d = idx & 63;
  int l = (idx >> 6) & (LL - 1);
  int s = idx >> 18;
  int dir = s & 3, b = s >> 2;
  int h = l >> 6, w = l & 63;
  int p;
  if      (dir == 0) p = (h << 6) | w;                // identity
  else if (dir == 1) p = (w << 6) | h;                // transpose
  else if (dir == 2) p = (h << 6) | ((h + w) & 63);   // diag gather
  else               p = (h << 6) | ((w - h) & 63);   // antidiag gather
  float v = xc[(((size_t)b * CHN + dir * DD + d) << 12) + p];
  xs[idx] = (bf16)v;
}

// ---------------- K4: GEMM2  xz(s, j, l) = xs @ Win^T + bin ----------------
// per-wave 16(l) x 64(j) tile, K=64: one xs fragment shared by 4 WMMAs.
// Stores channel-major (j, l): 8 contiguous f32 per lane per j-tile.
__global__ void k_gemm2(const bf16* __restrict__ xs, const bf16* __restrict__ winb,
                        const float* __restrict__ bin, float* __restrict__ xz) {
  int wave = (blockIdx.x * blockDim.x + threadIdx.x) >> 5;
  int lane = threadIdx.x & 31;
  int jt = wave & 3;
  int lt = (wave >> 2) & 255;
  int s  = wave >> 10;
  int j0 = jt * 64, l0 = lt * 16;
  const bf16* Arow = xs + ((size_t)s * LL) * DD;
  int n = lane & 15, hs = lane >> 4;
  v8f acc[4];
  #pragma unroll
  for (int t = 0; t < 4; t++) {
    float cinit = bin[j0 + t * 16 + n];
    #pragma unroll
    for (int v = 0; v < 8; v++) acc[t][v] = cinit;
  }
  #pragma unroll
  for (int k = 0; k < DD; k += 32) {
    v16bf a = load_fragA(Arow, DD, l0, k, lane);
    #pragma unroll
    for (int t = 0; t < 4; t++) {
      v16bf bb = load_fragB(winb, DD, j0 + t * 16, k, lane);
      acc[t] = wmma_bf16(a, bb, acc[t]);
    }
  }
  #pragma unroll
  for (int t = 0; t < 4; t++) {
    float* dst = xz + (((size_t)s * JZ + j0 + t * 16 + n) << 12) + l0 + 8 * hs;
    *(v8f*)dst = acc[t];
  }
}

// ---------------- K5: depthwise causal conv1d (K=4) + SiLU ----------------
// reads xz channels [0,128); writes xiA (s,e,l) f32 and xiT (s,l,e) bf16
__global__ void k_conv1d_silu(const float* __restrict__ xz, const float* __restrict__ cw,
                              const float* __restrict__ cb, float* __restrict__ xiA,
                              bf16* __restrict__ xiT) {
  int idx = blockIdx.x * blockDim.x + threadIdx.x;  // (s, e, l), l fastest
  int l = idx & (LL - 1);
  int e = (idx >> 12) & 127;
  int s = idx >> 19;
  const float* src = xz + (((size_t)s * JZ + e) << 12);
  float acc = cb[e];
  #pragma unroll
  for (int k = 0; k < 4; k++) {
    int li = l - 3 + k;
    if (li >= 0) acc += cw[e * 4 + k] * src[li];
  }
  float v = siluf_(acc);
  xiA[(((size_t)s * EE + e) << 12) + l] = v;
  xiT[(((size_t)s * LL + l) << 7) + e]  = (bf16)v;
}

// ---------------- K6: GEMM3  dbl(s, l, 48) = xiT @ Wx^T (padded) ----------------
// per-wave 16(l) x 48(j) tile, K=128: one xiT fragment shared by 3 WMMAs.
__global__ void k_gemm3(const bf16* __restrict__ xiT, const bf16* __restrict__ wxpb,
                        float* __restrict__ dbl) {
  int wave = (blockIdx.x * blockDim.x + threadIdx.x) >> 5;
  int lane = threadIdx.x & 31;
  int lt = wave & 255;
  int s  = wave >> 8;
  int l0 = lt * 16;
  const bf16* Arow = xiT + ((size_t)s * LL) * EE;
  v8f acc[3] = {};
  #pragma unroll
  for (int k = 0; k < EE; k += 32) {
    v16bf a = load_fragA(Arow, EE, l0, k, lane);
    #pragma unroll
    for (int t = 0; t < 3; t++) {
      v16bf bb = load_fragB(wxpb, EE, t * 16, k, lane);
      acc[t] = wmma_bf16(a, bb, acc[t]);
    }
  }
  int n = lane & 15, hs = lane >> 4;
  #pragma unroll
  for (int t = 0; t < 3; t++) {
    #pragma unroll
    for (int v = 0; v < 8; v++) {
      int lrow = l0 + v + 8 * hs;
      dbl[((size_t)s * LL + lrow) * JX + t * 16 + n] = acc[t][v];
    }
  }
}

// ---------------- K7: selective scan ----------------
// 16 lanes per (s,e) channel; lane n carries state h_n.  y(s,l,e) bf16 out.
__global__ void k_scan(const float* __restrict__ dbl, const float* __restrict__ xiA,
                       const float* __restrict__ xz, const float* __restrict__ Wdt,
                       const float* __restrict__ bdt, const float* __restrict__ Alog,
                       const float* __restrict__ Dskip, bf16* __restrict__ y) {
  int tid = blockIdx.x * blockDim.x + threadIdx.x;
  int ch  = tid >> 4;          // 4096 channels
  int n   = tid & 15;
  int s = ch >> 7;
  int e = ch & 127;
  float a  = -__expf(Alog[e * NN + n]);
  float w0 = Wdt[e * 4 + 0], w1 = Wdt[e * 4 + 1], w2 = Wdt[e * 4 + 2], w3 = Wdt[e * 4 + 3];
  float bde = bdt[e];
  float dsk = Dskip[e];
  const float* xi_c = xiA + (((size_t)s * EE + e) << 12);
  const float* z_c  = xz  + (((size_t)s * JZ + EE + e) << 12);
  const float* drow = dbl + ((size_t)s * LL) * JX;
  bf16* yrow = y + ((size_t)s * LL) * EE + e;
  float h = 0.f;
  for (int l = 0; l < LL; l++) {
    const float* dl = drow + (size_t)l * JX;
    float dt = softplusf_(w0 * dl[0] + w1 * dl[1] + w2 * dl[2] + w3 * dl[3] + bde);
    float Bn = dl[4 + n];
    float Cn = dl[20 + n];
    float xv = xi_c[l];
    h = h * __expf(dt * a) + (dt * xv) * Bn;
    float part = h * Cn;
    // reduce over 16-lane group (wave32 xor shuffles stay inside the group)
    part += __shfl_xor(part, 8, 32);
    part += __shfl_xor(part, 4, 32);
    part += __shfl_xor(part, 2, 32);
    part += __shfl_xor(part, 1, 32);
    if (n == 0) {
      float zv = z_c[l];
      float yv = (part + dsk * xv) * siluf_(zv);
      yrow[(size_t)l * EE] = (bf16)yv;
    }
  }
}

// ---------------- K8: GEMM4  m(s, d, l) = y @ Wout^T + bout ----------------
// per-wave 16(l) x 64(d) tile, K=128: one y fragment shared by 4 WMMAs.
__global__ void k_gemm4(const bf16* __restrict__ y, const bf16* __restrict__ woutb,
                        const float* __restrict__ bout, float* __restrict__ m) {
  int wave = (blockIdx.x * blockDim.x + threadIdx.x) >> 5;
  int lane = threadIdx.x & 31;
  int lt = wave & 255;
  int s  = wave >> 8;
  int l0 = lt * 16;
  const bf16* Arow = y + ((size_t)s * LL) * EE;
  int n = lane & 15, hs = lane >> 4;
  v8f acc[4];
  #pragma unroll
  for (int t = 0; t < 4; t++) {
    float cinit = bout[t * 16 + n];
    #pragma unroll
    for (int v = 0; v < 8; v++) acc[t][v] = cinit;
  }
  #pragma unroll
  for (int k = 0; k < EE; k += 32) {
    v16bf a = load_fragA(Arow, EE, l0, k, lane);
    #pragma unroll
    for (int t = 0; t < 4; t++) {
      v16bf bb = load_fragB(woutb, EE, t * 16, k, lane);
      acc[t] = wmma_bf16(a, bb, acc[t]);
    }
  }
  #pragma unroll
  for (int t = 0; t < 4; t++) {
    float* dst = m + (((size_t)s * DD + t * 16 + n) << 12) + l0 + 8 * hs;
    *(v8f*)dst = acc[t];
  }
}

// ---------------- K9: scatter mamba outputs into d_out (initializes d_out) ----------
__global__ void k_scatter_m(const float* __restrict__ m, float* __restrict__ out) {
  int idx = blockIdx.x * blockDim.x + threadIdx.x;  // (b, g, d, h, w)
  int w = idx & 63;
  int h = (idx >> 6) & 63;
  int d = (idx >> 12) & 63;
  int g = (idx >> 18) & 3;
  int b = idx >> 20;
  int l;
  if      (g == 0) l = (h << 6) | w;                    // identity
  else if (g == 1) l = (w << 6) | h;                    // transpose back
  else if (g == 2) l = (((w - h) & 63) << 6) | h;       // diag scatter
  else             l = (((h + w) & 63) << 6) | h;       // antidiag scatter
  out[idx] = m[(((size_t)(4 * b + g) * DD + d) << 12) + l];
}

// ---------------- K10: horizontal 1x5 conv (group 0), LDS row staging ----------------
__global__ void k_c1(const float* __restrict__ xc, const float* __restrict__ wgt,
                     const float* __restrict__ bias, float* __restrict__ out) {
  __shared__ float row[68];
  int b = blockIdx.x >> 6;
  int h = blockIdx.x & 63;
  int t = threadIdx.x;
  int d  = t & 63;
  int wg = t >> 6;
  float acc[16];
  #pragma unroll
  for (int i = 0; i < 16; i++) acc[i] = 0.f;
  const float* plane = xc + ((size_t)b * CHN << 12) + (h << 6);
  for (int ci = 0; ci < CHN; ci++) {
    if (t < 68) {
      int w = t - 2;
      row[t] = ((unsigned)w < 64u) ? plane[((size_t)ci << 12) + w] : 0.f;
    }
    __syncthreads();
    const float* wp = wgt + ((size_t)d * CHN + ci) * 5;
    float c0 = wp[0], c1 = wp[1], c2 = wp[2], c3 = wp[3], c4 = wp[4];
    #pragma unroll
    for (int i = 0; i < 16; i++) {
      int w = wg * 16 + i;
      acc[i] += c0 * row[w] + c1 * row[w + 1] + c2 * row[w + 2] + c3 * row[w + 3] + c4 * row[w + 4];
    }
    __syncthreads();
  }
  float bia = bias[d];
  #pragma unroll
  for (int i = 0; i < 16; i++) {
    int w = wg * 16 + i;
    float* o = out + (((size_t)b * CHN + d) << 12) + (h << 6) + w;
    *o += acc[i] + bia;
  }
}

// ---------------- K11: vertical 5x1 conv (group 1), LDS 5-row staging ----------------
__global__ void k_c2(const float* __restrict__ xc, const float* __restrict__ wgt,
                     const float* __restrict__ bias, float* __restrict__ out) {
  __shared__ float cols[320];
  int b = blockIdx.x >> 6;
  int h = blockIdx.x & 63;
  int t = threadIdx.x;
  int d  = t & 63;
  int wg = t >> 6;
  float acc[16];
  #pragma unroll
  for (int i = 0; i < 16; i++) acc[i] = 0.f;
  const float* plane = xc + ((size_t)b * CHN << 12);
  for (int ci = 0; ci < CHN; ci++) {
    for (int i = t; i < 320; i += 256) {
      int tt = i >> 6;
      int w  = i & 63;
      int hh = h - 2 + tt;
      cols[i] = ((unsigned)hh < 64u) ? plane[((size_t)ci << 12) + (hh << 6) + w] : 0.f;
    }
    __syncthreads();
    const float* wp = wgt + ((size_t)d * CHN + ci) * 5;
    float c0 = wp[0], c1 = wp[1], c2 = wp[2], c3 = wp[3], c4 = wp[4];
    #pragma unroll
    for (int i = 0; i < 16; i++) {
      int w = wg * 16 + i;
      acc[i] += c0 * cols[w] + c1 * cols[64 + w] + c2 * cols[128 + w] +
                c3 * cols[192 + w] + c4 * cols[256 + w];
    }
    __syncthreads();
  }
  float bia = bias[d];
  #pragma unroll
  for (int i = 0; i < 16; i++) {
    int w = wg * 16 + i;
    float* o = out + (((size_t)b * CHN + DD + d) << 12) + (h << 6) + w;
    *o += acc[i] + bia;
  }
}

// ---------------- K12: diag-shifted conv (group 2) ----------------
__global__ void k_c3(const float* __restrict__ xc, const float* __restrict__ wgt,
                     const float* __restrict__ bias, float* __restrict__ out) {
  int idx = blockIdx.x * blockDim.x + threadIdx.x;  // (b, d, h, w)
  int w = idx & 63, h = (idx >> 6) & 63, d = (idx >> 12) & 63, b = idx >> 18;
  int k = (h << 7) + w;
  int q = k / 127, r = k - 127 * q;
  int   off[5];
  float msk[5];
  #pragma unroll
  for (int t = 0; t < 5; t++) {
    int qp = q - 2 + t;
    int qc = qp < 0 ? 0 : (qp > 63 ? 63 : qp);
    int mm = qc * 127 + r;
    int hh = mm >> 7, wp = mm & 127;
    bool ok = ((unsigned)qp < 64u) && (wp < 64);
    off[t] = ok ? ((hh << 6) + wp) : 0;
    msk[t] = ok ? 1.f : 0.f;
  }
  float acc = 0.f;
  const float* plane = xc + ((size_t)b * CHN << 12);
  for (int ci = 0; ci < CHN; ci++) {
    const float* src = plane + ((size_t)ci << 12);
    const float* wpv = wgt + ((size_t)d * CHN + ci) * 5;
    #pragma unroll
    for (int t = 0; t < 5; t++) acc += wpv[t] * msk[t] * src[off[t]];
  }
  float* o = out + (((size_t)b * CHN + 2 * DD + d) << 12) + (h << 6) + w;
  *o += acc + bias[d];
}

// ---------------- K13: antidiag-shifted conv (group 3) ----------------
__global__ void k_c4(const float* __restrict__ xc, const float* __restrict__ wgt,
                     const float* __restrict__ bias, float* __restrict__ out) {
  int idx = blockIdx.x * blockDim.x + threadIdx.x;  // (b, d, h, w)
  int w = idx & 63, h = (idx >> 6) & 63, d = (idx >> 12) & 63, b = idx >> 18;
  int k = (w << 7) + h;
  int q = k % 127, r = k / 127;
  int   off[5];
  float msk[5];
  #pragma unroll
  for (int t = 0; t < 5; t++) {
    int rp = r - 2 + t;
    int rc = rp < 0 ? 0 : (rp > 63 ? 63 : rp);
    int mm = rc * 127 + q;
    int ww = mm >> 7, hh = mm & 127;
    bool ok = ((unsigned)rp < 64u) && (hh < 64);
    off[t] = ok ? ((hh << 6) + ww) : 0;
    msk[t] = ok ? 1.f : 0.f;
  }
  float acc = 0.f;
  const float* plane = xc + ((size_t)b * CHN << 12);
  for (int ci = 0; ci < CHN; ci++) {
    const float* src = plane + ((size_t)ci << 12);
    const float* wpv = wgt + ((size_t)d * CHN + ci) * 5;
    #pragma unroll
    for (int t = 0; t < 5; t++) acc += wpv[t] * msk[t] * src[off[t]];
  }
  float* o = out + (((size_t)b * CHN + 3 * DD + d) << 12) + (h << 6) + w;
  *o += acc + bias[d];
}

// ---------------- workspace layout (bytes); total ~412 MB ----------------
static constexpr size_t OFF_W1   = 0;                                // 262144
static constexpr size_t OFF_WIN  = OFF_W1   + 262144;                // 32768
static constexpr size_t OFF_WXP  = OFF_WIN  + 32768;                 // 12288
static constexpr size_t OFF_WOUT = OFF_WXP  + 12288;                 // 16384
static constexpr size_t OFF_CAT  = OFF_WOUT + 16384;                 // 33554432 (b,p,ci) bf16
static constexpr size_t OFF_XC   = OFF_CAT  + 33554432;              // 33554432 (b,c,p) f32
static constexpr size_t OFF_XS   = OFF_XC   + 33554432;              // 16777216 (s,l,d) bf16
static constexpr size_t OFF_XZ   = OFF_XS   + 16777216;              // 134217728 (s,j,l) f32
static constexpr size_t OFF_XIA  = OFF_XZ   + 134217728;             // 67108864 (s,e,l) f32
static constexpr size_t OFF_XIT  = OFF_XIA  + 67108864;              // 33554432 (s,l,e) bf16
static constexpr size_t OFF_DBL  = OFF_XIT  + 33554432;              // 25165824 (s,l,48) f32
static constexpr size_t OFF_Y    = OFF_DBL  + 25165824;              // 33554432 (s,l,e) bf16
static constexpr size_t OFF_M    = OFF_Y    + 33554432;              // 33554432 (s,d,l) f32

extern "C" void kernel_launch(void* const* d_in, const int* in_sizes, int n_in,
                              void* d_out, int out_size, void* d_ws, size_t ws_size,
                              hipStream_t stream) {
  const float* x      = (const float*)d_in[0];
  const float* x2     = (const float*)d_in[1];
  const float* w1     = (const float*)d_in[2];
  const float* b1     = (const float*)d_in[3];
  const float* bn_g   = (const float*)d_in[4];
  const float* bn_b   = (const float*)d_in[5];
  const float* bn_m   = (const float*)d_in[6];
  const float* bn_v   = (const float*)d_in[7];
  const float* hconv_w= (const float*)d_in[8];
  const float* hconv_b= (const float*)d_in[9];
  const float* wconv_w= (const float*)d_in[10];
  const float* wconv_b= (const float*)d_in[11];
  const float* d19_w  = (const float*)d_in[12];
  const float* d19_b  = (const float*)d_in[13];
  const float* d37_w  = (const float*)d_in[14];
  const float* d37_b  = (const float*)d_in[15];
  const float* Win    = (const float*)d_in[16];
  const float* bin_   = (const float*)d_in[17];
  const float* convw  = (const float*)d_in[18];
  const float* convb  = (const float*)d_in[19];
  const float* Wx     = (const float*)d_in[20];
  const float* Wdt    = (const float*)d_in[21];
  const float* bdt    = (const float*)d_in[22];
  const float* Alog   = (const float*)d_in[23];
  const float* Dskip  = (const float*)d_in[24];
  const float* Wout   = (const float*)d_in[25];
  const float* bout   = (const float*)d_in[26];
  float* out = (float*)d_out;

  char* ws = (char*)d_ws;
  bf16*  w1b   = (bf16*)(ws + OFF_W1);
  bf16*  winb  = (bf16*)(ws + OFF_WIN);
  bf16*  wxpb  = (bf16*)(ws + OFF_WXP);
  bf16*  woutb = (bf16*)(ws + OFF_WOUT);
  bf16*  cat   = (bf16*)(ws + OFF_CAT);
  float* xc    = (float*)(ws + OFF_XC);
  bf16*  xs    = (bf16*)(ws + OFF_XS);
  float* xz    = (float*)(ws + OFF_XZ);
  float* xiA   = (float*)(ws + OFF_XIA);
  bf16*  xiT   = (bf16*)(ws + OFF_XIT);
  float* dbl   = (float*)(ws + OFF_DBL);
  bf16*  y     = (bf16*)(ws + OFF_Y);
  float* m     = (float*)(ws + OFF_M);

  const int TB = 256;
  k_prep_weights<<<512, TB, 0, stream>>>(w1, Win, Wx, Wout, w1b, winb, wxpb, woutb);
  k_cat_transpose<<<65536, TB, 0, stream>>>(x, x2, cat);
  k_gemm1<<<1024, TB, 0, stream>>>(w1b, cat, b1, bn_g, bn_b, bn_m, bn_v, xc);
  k_rearrange<<<32768, TB, 0, stream>>>(xc, xs);
  k_gemm2<<<4096, TB, 0, stream>>>(xs, winb, bin_, xz);
  k_conv1d_silu<<<65536, TB, 0, stream>>>(xz, convw, convb, xiA, xiT);
  k_gemm3<<<1024, TB, 0, stream>>>(xiT, wxpb, dbl);
  k_scan<<<256, TB, 0, stream>>>(dbl, xiA, xz, Wdt, bdt, Alog, Dskip, y);
  k_gemm4<<<1024, TB, 0, stream>>>(y, woutb, bout, m);
  k_scatter_m<<<32768, TB, 0, stream>>>(m, out);
  k_c1<<<512, TB, 0, stream>>>(xc, hconv_w, hconv_b, out);
  k_c2<<<512, TB, 0, stream>>>(xc, wconv_w, wconv_b, out);
  k_c3<<<8192, TB, 0, stream>>>(xc, d19_w, d19_b, out);
  k_c4<<<8192, TB, 0, stream>>>(xc, d37_w, d37_b, out);
}
